// TokenChoiceMoE_21938692948542
// MI455X (gfx1250) — matrix operand order repaired and checked by reading, not compiled, not run on hardware
//
#include <hip/hip_runtime.h>
#include <hip/hip_bf16.h>

// ---------------- problem constants (from reference) ----------------
#define NTOK      16384      // B*S = 4*4096
#define HDIM      1024
#define FFN_DIM   4096
#define NEXP      16
#define EMLP      12         // E - (1 zero + 1 copy + 2 const)
#define MLPSTART  4
#define CAP       2048       // ceil(1.0 * NTOK*2 / 16)
#define EPSV      1e-5f

typedef __attribute__((ext_vector_type(16))) __bf16 v16bf;
typedef __attribute__((ext_vector_type(8)))  float  v8f;

// =====================================================================
// Kernel 0: fp32 -> bf16 weight conversion (streaming, once per launch)
// =====================================================================
__global__ void cvt_bf16_kernel(const float4* __restrict__ in,
                                uint2* __restrict__ out, long n4) {
  long i      = (long)blockIdx.x * blockDim.x + threadIdx.x;
  long stride = (long)gridDim.x * blockDim.x;
  for (; i < n4; i += stride) {
    float4 v = in[i];
    __bf16 a = (__bf16)v.x, b = (__bf16)v.y, c = (__bf16)v.z, d = (__bf16)v.w;
    unsigned lo = ((unsigned)__builtin_bit_cast(unsigned short, b) << 16) |
                  (unsigned)__builtin_bit_cast(unsigned short, a);
    unsigned hi = ((unsigned)__builtin_bit_cast(unsigned short, d) << 16) |
                  (unsigned)__builtin_bit_cast(unsigned short, c);
    out[i] = make_uint2(lo, hi);
  }
}

// =====================================================================
// Kernel 1: RMSNorm + router logits + softmax + top-2   (1 block/token)
// =====================================================================
__global__ void rmsnorm_router_kernel(const float* __restrict__ x,
                                      const float* __restrict__ nw,
                                      const float* __restrict__ rw,
                                      const float* __restrict__ rb,
                                      float* __restrict__ xn,
                                      int*   __restrict__ eidx,
                                      float* __restrict__ gval) {
  __shared__ float xns[HDIM];
  __shared__ float red[256];
  __shared__ float lg[NEXP];
  const int  t   = threadIdx.x;
  const long tok = blockIdx.x;
  const float* xr = x + tok * HDIM;

  float ss = 0.f;
  #pragma unroll
  for (int j = 0; j < 4; ++j) { float v = xr[t + j*256]; ss += v*v; }
  red[t] = ss;
  __syncthreads();
  for (int s = 128; s > 0; s >>= 1) {
    if (t < s) red[t] += red[t + s];
    __syncthreads();
  }
  const float rs = rsqrtf(red[0] / (float)HDIM + EPSV);
  __syncthreads();                       // before red[] reuse

  #pragma unroll
  for (int j = 0; j < 4; ++j) {
    int h = t + j*256;
    float v = xr[h] * rs * nw[h];
    xns[h] = v;
    xn[tok*HDIM + h] = v;
  }
  __syncthreads();

  // 16 experts x 16 segments of 64
  const int e   = t >> 4;
  const int seg = t & 15;
  const float* wrow = rw + (long)e*HDIM + seg*64;
  const float* xsg  = xns + seg*64;
  float p = 0.f;
  #pragma unroll 8
  for (int j = 0; j < 64; ++j) p += xsg[j] * wrow[j];
  red[t] = p;
  __syncthreads();
  if (t < NEXP) {
    float l = rb[t];
    #pragma unroll
    for (int s2 = 0; s2 < 16; ++s2) l += red[t*16 + s2];
    lg[t] = l;
  }
  __syncthreads();
  if (t == 0) {
    float m = lg[0];
    for (int i = 1; i < NEXP; ++i) m = fmaxf(m, lg[i]);
    float den = 0.f;
    for (int i = 0; i < NEXP; ++i) den += __expf(lg[i] - m);
    int b0 = 0; float v0 = lg[0];
    for (int i = 1; i < NEXP; ++i) if (lg[i] > v0) { v0 = lg[i]; b0 = i; }
    int b1 = -1; float v1 = -3.0e38f;
    for (int i = 0; i < NEXP; ++i) if (i != b0 && lg[i] > v1) { v1 = lg[i]; b1 = i; }
    eidx[tok*2 + 0] = b0;  eidx[tok*2 + 1] = b1;
    gval[tok*2 + 0] = __expf(v0 - m) / den;
    gval[tok*2 + 1] = __expf(v1 - m) / den;
  }
}

// =====================================================================
// Kernel 2: ordered capacity assignment (matches reference cumsum order)
// =====================================================================
__global__ void routing_scan_kernel(const int*   __restrict__ eidx,
                                    const float* __restrict__ gval,
                                    int*   __restrict__ slot_arr,
                                    float* __restrict__ w_arr,
                                    int*   __restrict__ slot_map) {
  __shared__ int sh[512 * NEXP];       // 32 KB
  const int t = threadIdx.x;
  for (int i = t; i < NEXP*CAP; i += 512) slot_map[i] = -1;
  #pragma unroll
  for (int e = 0; e < NEXP; ++e) sh[t*NEXP + e] = 0;
  __syncthreads();

  const int PER  = (NTOK * 2) / 512;   // 64
  const int base = t * PER;
  for (int j = 0; j < PER; ++j) sh[t*NEXP + eidx[base + j]]++;
  __syncthreads();

  if (t < NEXP) {                      // exclusive scan per expert (in order)
    int run = 0;
    for (int b = 0; b < 512; ++b) {
      int c = sh[b*NEXP + t];
      sh[b*NEXP + t] = run;
      run += c;
    }
  }
  __syncthreads();

  for (int j = 0; j < PER; ++j) {
    int i   = base + j;
    int e   = eidx[i];
    int pos = sh[t*NEXP + e]++;
    int keep = (pos < CAP) ? 1 : 0;
    int slot = keep ? pos : (CAP - 1);
    slot_arr[i] = slot;
    w_arr[i]    = keep ? gval[i] : 0.f;
    if (keep) slot_map[e*CAP + slot] = (i >> 1);     // token index
  }
}

// =====================================================================
// Kernel 3: dispatch — expert 1 rows fp32, MLP expert rows bf16
// =====================================================================
__global__ void dispatch_kernel(const float* __restrict__ xn,
                                const int*   __restrict__ slot_map,
                                float*  __restrict__ disp_copy,
                                __bf16* __restrict__ disp_bf) {
  const int slot = blockIdx.x;
  const int eg   = blockIdx.y;
  const int e    = (eg == 0) ? 1 : (eg + 3);
  const int tok  = slot_map[e*CAP + slot];
  const int t    = threadIdx.x;
  const float* src = xn + (long)(tok < 0 ? 0 : tok) * HDIM;
  if (e == 1) {
    float* dst = disp_copy + (long)slot * HDIM;
    #pragma unroll
    for (int j = 0; j < 4; ++j) { int h = t + 256*j; dst[h] = (tok >= 0) ? src[h] : 0.f; }
  } else {
    __bf16* dst = disp_bf + ((long)(e - MLPSTART)*CAP + slot) * HDIM;
    #pragma unroll
    for (int j = 0; j < 4; ++j) { int h = t + 256*j; dst[h] = (tok >= 0) ? (__bf16)src[h] : (__bf16)0.f; }
  }
}

// =====================================================================
// Kernel 4/5: batched expert GEMM, D = A(bf16, MxK) * W(bf16, NxK)^T + b
//   Block tile 128x128x32, 8 waves (2M x 4N), wave tile 64x32 (4x2 accs
//   = 64 VGPRs -> spill-free under the 256-VGPR window).
//   BOTH tiles staged with CDNA5 async global->LDS DMA (ASYNCcnt),
//   double-buffered: next tile's HBM/L2 traffic overlaps current WMMA.
//   mode 0: GELU epilogue, bf16 out (fc1) ; mode 1: fp32 out (fc2)
// =====================================================================
#define BM 128
#define BN 128
#define BK 32
#define LDS_S 40     // padded row stride in elements (80B: 16B-aligned, bank-skewed)

union FragBF { v16bf v; uint4 q[2]; };

// async copy of one 128x32 bf16 tile: thread t moves 32B (2 x b128)
__device__ __forceinline__ void stage_tile_async(const __bf16* __restrict__ base, int K,
                                                 int r0, int kt,
                                                 unsigned short* dstBase, int t) {
  const int r    = t >> 1;
  const int half = t & 1;
  const __bf16* g = base + (size_t)(r0 + r) * K + kt + half * 16;
  unsigned long long ga = (unsigned long long)(uintptr_t)g;
  unsigned lds = (unsigned)(uintptr_t)(dstBase + r * LDS_S + half * 16);
  asm volatile("global_load_async_to_lds_b128 %0, %1, off\n\t"
               "global_load_async_to_lds_b128 %0, %1, off offset:16"
               :: "v"(lds), "v"(ga) : "memory");
}

__global__ void moe_gemm_kernel(const __bf16* __restrict__ Aall,
                                const __bf16* __restrict__ Wall,
                                const float*  __restrict__ Ball,
                                void*         __restrict__ Oall,
                                int M, int Np, int K, int mode) {
  __shared__ __align__(16) unsigned short As[2][BM * LDS_S];   // 2 x 10 KB
  __shared__ __align__(16) unsigned short Bs[2][BN * LDS_S];   // 2 x 10 KB

  const int ex = blockIdx.z;
  const __bf16* A    = Aall + (size_t)ex * M * K;
  const __bf16* W    = Wall + (size_t)ex * Np * K;
  const float*  bias = Ball + (size_t)ex * Np;

  const int n0 = blockIdx.x * BN;
  const int m0 = blockIdx.y * BM;
  const int t     = threadIdx.x;
  const int lane  = t & 31;
  const int wid   = t >> 5;
  const int waveN = wid & 3;    // 4 waves across N (32 each)
  const int waveM = wid >> 2;   // 2 waves across M (64 each)

  const v8f vzero = {0.f,0.f,0.f,0.f,0.f,0.f,0.f,0.f};
  v8f acc[4][2];
  #pragma unroll
  for (int s = 0; s < 4; ++s)
    #pragma unroll
    for (int u = 0; u < 2; ++u) acc[s][u] = vzero;

  const int am = lane & 15;
  const int kg = lane >> 4;

  // ---- prologue: stage tile 0 into buffer 0 ----
  stage_tile_async(A, K, m0, 0, &As[0][0], t);
  stage_tile_async(W, K, n0, 0, &Bs[0][0], t);
  asm volatile("s_wait_asynccnt 0" ::: "memory");
  __syncthreads();

  int buf = 0;
  for (int kt = 0; kt < K; kt += BK) {
    const int  ktn  = kt + BK;
    const bool more = ktn < K;
    if (more) {                         // overlap next tile's traffic with WMMA
      stage_tile_async(A, K, m0, ktn, &As[buf ^ 1][0], t);
      stage_tile_async(W, K, n0, ktn, &Bs[buf ^ 1][0], t);
    }

    // ---- fragment loads per ISA VGPR layouts ----
    FragBF af[4], bfb[2];
    #pragma unroll
    for (int s = 0; s < 4; ++s) {
      int row = waveM*64 + s*16 + am;
      af[s].q[0] = *reinterpret_cast<const uint4*>(&As[buf][row*LDS_S + kg*8]);       // K kg*8..+7
      af[s].q[1] = *reinterpret_cast<const uint4*>(&As[buf][row*LDS_S + 16 + kg*8]);  // K 16+kg*8..+7
    }
    #pragma unroll
    for (int u = 0; u < 2; ++u) {
      int col = waveN*32 + u*16 + am;
      bfb[u].q[0] = *reinterpret_cast<const uint4*>(&Bs[buf][col*LDS_S + kg*8]);       // K kg*8..+7
      bfb[u].q[1] = *reinterpret_cast<const uint4*>(&Bs[buf][col*LDS_S + 16 + kg*8]);  // K 16+kg*8..+7
    }

    #pragma unroll
    for (int s = 0; s < 4; ++s)
      #pragma unroll
      for (int u = 0; u < 2; ++u)
        acc[s][u] = __builtin_amdgcn_wmma_f32_16x16x32_bf16(
            false, af[s].v, false, bfb[u].v, (short)0, acc[s][u], false, false);

    asm volatile("s_wait_asynccnt 0" ::: "memory");
    __syncthreads();
    buf ^= 1;
  }

  // ---- epilogue: C/D layout M = vgpr + 8*(lane>=16), N = lane&15 ----
  const int cn  = lane & 15;
  const int cm8 = (lane >> 4) * 8;
  #pragma unroll
  for (int s = 0; s < 4; ++s) {
    #pragma unroll
    for (int u = 0; u < 2; ++u) {
      const int gmB = m0 + waveM*64 + s*16 + cm8;
      const int gn  = n0 + waveN*32 + u*16 + cn;
      const float bv = bias[gn];
      #pragma unroll
      for (int r = 0; r < 8; ++r) {
        float v = acc[s][u][r] + bv;
        size_t oi = (size_t)ex * M * Np + (size_t)(gmB + r) * Np + gn;
        if (mode == 0) {
          float g = 0.5f * v * (1.0f + erff(v * 0.70710678118654752f));  // exact GELU
          reinterpret_cast<__bf16*>(Oall)[oi] = (__bf16)g;
        } else {
          reinterpret_cast<float*>(Oall)[oi] = v;
        }
      }
    }
  }
}

// =====================================================================
// Kernel 6: combine — gather expert outputs, weight by gates, normalize
// =====================================================================
__device__ inline float fetch_val(int e, int slot, int h,
                                  const float* __restrict__ dc,
                                  const float* __restrict__ mo,
                                  const float* __restrict__ cv) {
  if (e == 0) return 0.f;
  if (e == 1) return dc[(long)slot*HDIM + h];
  if (e < MLPSTART) return cv[(long)(e - 2)*HDIM + h];
  return mo[((size_t)(e - MLPSTART)*CAP + slot)*HDIM + h];
}

__global__ void combine_kernel(const int*   __restrict__ eidx,
                               const int*   __restrict__ slot_arr,
                               const float* __restrict__ w_arr,
                               const float* __restrict__ disp_copy,
                               const float* __restrict__ mlp_out,
                               const float* __restrict__ cvec,
                               float* __restrict__ out) {
  const long tok = blockIdx.x;
  const int  t   = threadIdx.x;
  const int   e0 = eidx[tok*2],     e1 = eidx[tok*2 + 1];
  const int   s0 = slot_arr[tok*2], s1 = slot_arr[tok*2 + 1];
  const float w0 = w_arr[tok*2],    w1 = w_arr[tok*2 + 1];
  const float gs = w0 + w1;
  const float inv = 1.0f / (gs > 0.f ? gs : 1.0f);
  #pragma unroll
  for (int j = 0; j < 4; ++j) {
    int h = t + 256*j;
    float v0 = fetch_val(e0, s0, h, disp_copy, mlp_out, cvec);
    float v1 = fetch_val(e1, s1, h, disp_copy, mlp_out, cvec);
    out[tok*HDIM + h] = (w0*v0 + w1*v1) * inv;
  }
}

// =====================================================================
extern "C" void kernel_launch(void* const* d_in, const int* in_sizes, int n_in,
                              void* d_out, int out_size, void* d_ws, size_t ws_size,
                              hipStream_t stream) {
  const float* x    = (const float*)d_in[0];
  const float* nw   = (const float*)d_in[1];
  const float* rw   = (const float*)d_in[2];
  const float* rb   = (const float*)d_in[3];
  const float* fc1w = (const float*)d_in[4];
  const float* fc1b = (const float*)d_in[5];
  const float* fc2w = (const float*)d_in[6];
  const float* fc2b = (const float*)d_in[7];
  const float* cvec = (const float*)d_in[8];
  float* out = (float*)d_out;

  char* ws = (char*)d_ws;
  size_t off = 0;
  float*  xn        = (float*)(ws + off);  off += (size_t)NTOK*HDIM*sizeof(float);       //  64 MB
  float*  disp_copy = (float*)(ws + off);  off += (size_t)CAP*HDIM*sizeof(float);        //   8 MB
  __bf16* disp_bf   = (__bf16*)(ws + off); off += (size_t)EMLP*CAP*HDIM*2;               //  48 MB
  __bf16* h_bf      = (__bf16*)(ws + off); off += (size_t)EMLP*CAP*FFN_DIM*2;            // 192 MB
  float*  mlp_out   = (float*)(ws + off);  off += (size_t)EMLP*CAP*HDIM*sizeof(float);   //  96 MB
  __bf16* fc1w_bf   = (__bf16*)(ws + off); off += (size_t)EMLP*FFN_DIM*HDIM*2;           //  96 MB
  __bf16* fc2w_bf   = (__bf16*)(ws + off); off += (size_t)EMLP*HDIM*FFN_DIM*2;           //  96 MB
  int*    eidx      = (int*)(ws + off);    off += (size_t)NTOK*2*sizeof(int);
  float*  gval      = (float*)(ws + off);  off += (size_t)NTOK*2*sizeof(float);
  int*    slot_arr  = (int*)(ws + off);    off += (size_t)NTOK*2*sizeof(int);
  float*  w_arr     = (float*)(ws + off);  off += (size_t)NTOK*2*sizeof(float);
  int*    slot_map  = (int*)(ws + off);    off += (size_t)NEXP*CAP*sizeof(int);

  // weight precision conversion (streaming, overlaps nothing critical)
  const long n4 = (long)EMLP * FFN_DIM * HDIM / 4;    // 12.6M float4 per tensor
  cvt_bf16_kernel<<<8192, 256, 0, stream>>>((const float4*)fc1w, (uint2*)fc1w_bf, n4);
  cvt_bf16_kernel<<<8192, 256, 0, stream>>>((const float4*)fc2w, (uint2*)fc2w_bf, n4);

  rmsnorm_router_kernel<<<NTOK, 256, 0, stream>>>(x, nw, rw, rb, xn, eidx, gval);
  routing_scan_kernel<<<1, 512, 0, stream>>>(eidx, gval, slot_arr, w_arr, slot_map);
  dispatch_kernel<<<dim3(CAP, 13), 256, 0, stream>>>(xn, slot_map, disp_copy, disp_bf);

  // fc1 + GELU : A = dispatch bf16 (2048x1024), W = fc1 bf16 (4096x1024), out bf16
  moe_gemm_kernel<<<dim3(FFN_DIM/BN, CAP/BM, EMLP), 256, 0, stream>>>(
      disp_bf, fc1w_bf, fc1b, (void*)h_bf, CAP, FFN_DIM, HDIM, 0);
  // fc2 : A = h bf16 (2048x4096), W = fc2 bf16 (1024x4096), out fp32
  moe_gemm_kernel<<<dim3(HDIM/BN, CAP/BM, EMLP), 256, 0, stream>>>(
      h_bf, fc2w_bf, fc2b, (void*)mlp_out, CAP, HDIM, FFN_DIM, 1);

  combine_kernel<<<NTOK, 256, 0, stream>>>(eidx, slot_arr, w_arr, disp_copy, mlp_out, cvec, out);
}
